// EdgePooling_31825707664029
// MI455X (gfx1250) — matrix operand order repaired and checked by reading, not compiled, not run on hardware
//
#include <hip/hip_runtime.h>

typedef _Float16 v16h __attribute__((ext_vector_type(16)));
typedef float    v8f  __attribute__((ext_vector_type(8)));

#define IN_CH  128
#define OUT_CH 128
#define LDSW   136   // halves per LDS row of W_t (128 + 8 pad, keeps 16B alignment)

// ---------------------------------------------------------------------------
// Convert W_t (OUT_CH x IN_CH, f32 row-major) to f16 in LDS, padded rows.
// ---------------------------------------------------------------------------
__device__ __forceinline__ void load_Wt_to_lds(_Float16* sW, const float* __restrict__ W_t) {
    struct h2 { _Float16 x, y; };
    const float2* W2 = (const float2*)W_t;
    for (int i = threadIdx.x; i < (OUT_CH * IN_CH) / 2; i += 256) {
        int row = i >> 6;       // 64 float2 per row
        int cp  = i & 63;
        float2 w = W2[i];
        h2 v; v.x = (_Float16)w.x; v.y = (_Float16)w.y;
        *(h2*)(sW + row * LDSW + cp * 2) = v;
    }
}

// ---------------------------------------------------------------------------
// One wave: 16 rows x 128 out channels via 32x v_wmma_f32_16x16x32_f16.
// SELF==false: rows are edges, merged = x[src]+x[dst], also emits sigmoid score.
// SELF==true : rows are nodes, merged = 2*x[node].
// ---------------------------------------------------------------------------
template <bool SELF>
__device__ __forceinline__ void tile_compute(
    const float* __restrict__ x,
    const int*  __restrict__ srcIdx,     // null for SELF
    const int*  __restrict__ dstIdx,     // null for SELF
    const float* __restrict__ b_t,
    const float* __restrict__ W_s,
    float bsv,
    const _Float16* sW,
    float* __restrict__ outRows,
    float* __restrict__ scores,          // null for SELF
    long long rowBase)
{
    const int lane = threadIdx.x & 31;
    const int n    = lane & 15;   // column-in-tile / row-in-tile id
    const int sel  = lane >> 4;   // lane-half selector

    long long r = rowBase + n;
    const float* xs;
    const float* xd;
    if (SELF) {
        xs = x + r * (long long)IN_CH;
        xd = xs;
    } else {
        long long s = (long long)srcIdx[r];
        long long d = (long long)dstIdx[r];
        xs = x + s * (long long)IN_CH;
        xd = x + d * (long long)IN_CH;
    }

    // Build A (16x32 f16) for 4 K-chunks.
    // Lane m, sel=0: K = kc*32 + {0..7, 16..23}; sel=1: +8 offset (ISA layout).
    v16h A[4];
    #pragma unroll
    for (int kc = 0; kc < 4; ++kc) {
        #pragma unroll
        for (int g = 0; g < 2; ++g) {
            const int base = kc * 32 + sel * 8 + g * 16;
            const float4* ps = (const float4*)(xs + base);
            const float4* pd = (const float4*)(xd + base);
            float4 s0 = ps[0], s1 = ps[1];
            float4 d0, d1;
            if (SELF) { d0 = s0; d1 = s1; } else { d0 = pd[0]; d1 = pd[1]; }
            const int h = g * 8;
            A[kc][h + 0] = (_Float16)(s0.x + d0.x);
            A[kc][h + 1] = (_Float16)(s0.y + d0.y);
            A[kc][h + 2] = (_Float16)(s0.z + d0.z);
            A[kc][h + 3] = (_Float16)(s0.w + d0.w);
            A[kc][h + 4] = (_Float16)(s1.x + d1.x);
            A[kc][h + 5] = (_Float16)(s1.y + d1.y);
            A[kc][h + 6] = (_Float16)(s1.z + d1.z);
            A[kc][h + 7] = (_Float16)(s1.w + d1.w);
        }
    }

    float sacc[8];
    if (!SELF) {
        #pragma unroll
        for (int rr = 0; rr < 8; ++rr) sacc[rr] = 0.0f;
    }

    // 8 output tiles of 16 channels each.
    #pragma unroll
    for (int ot = 0; ot < 8; ++ot) {
        const int o0 = ot * 16;
        v8f c = {};
        #pragma unroll
        for (int kc = 0; kc < 4; ++kc) {
            // B (32x16 f16): lane n = column N=o0+n; halves = K = kc*32 + sel*8 + {0..7, 16..23}
            union { v16h v; float4 f[2]; } bu;
            const int off = (o0 + n) * LDSW + kc * 32 + sel * 8;
            bu.f[0] = *(const float4*)(sW + off);
            bu.f[1] = *(const float4*)(sW + off + 16);
            c = __builtin_amdgcn_wmma_f32_16x16x32_f16(
                    false, A[kc], false, bu.v, (short)0, c, false, false);
        }

        const float btv = b_t[o0 + n];
        float wsv = 0.0f;
        if (!SELF) wsv = W_s[o0 + n];

        // C/D layout: vgpr rr -> row rowBase + sel*8 + rr, col o0 + n
        #pragma unroll
        for (int rr = 0; rr < 8; ++rr) {
            const float v = c[rr] + btv;
            outRows[(rowBase + sel * 8 + rr) * (long long)OUT_CH + (o0 + n)] = v;
            if (!SELF) sacc[rr] += v * wsv;
        }
    }

    if (!SELF) {
        // Reduce the 128-channel dot across the 16 lanes of each half.
        #pragma unroll
        for (int rr = 0; rr < 8; ++rr) {
            float v = sacc[rr];
            v += __shfl_xor(v, 1, 32);
            v += __shfl_xor(v, 2, 32);
            v += __shfl_xor(v, 4, 32);
            v += __shfl_xor(v, 8, 32);
            sacc[rr] = v;
        }
        if (n == 0) {
            #pragma unroll
            for (int rr = 0; rr < 8; ++rr) {
                const float raw = sacc[rr] + bsv;
                scores[rowBase + sel * 8 + rr] = 1.0f / (1.0f + __expf(-raw));
            }
        }
    }
}

// ---------------------------------------------------------------------------
__global__ __launch_bounds__(256) void edgepool_edge_kernel(
    const float* __restrict__ x,
    const int*  __restrict__ ei,      // [2, E]
    long long E, long long tiles,
    const float* __restrict__ W_t,
    const float* __restrict__ b_t,
    const float* __restrict__ W_s,
    const float* __restrict__ b_s,
    float* __restrict__ x_merged,
    float* __restrict__ scores)
{
    __shared__ _Float16 sW[OUT_CH * LDSW];
    load_Wt_to_lds(sW, W_t);
    __syncthreads();

    const int wave = threadIdx.x >> 5;
    const long long tile = (long long)blockIdx.x * 8 + wave;
    if (tile >= tiles) return;   // wave-uniform: WMMA waves keep EXEC all-ones

    tile_compute<false>(x, ei, ei + E, b_t, W_s, b_s[0], sW,
                        x_merged, scores, tile * 16);
}

__global__ __launch_bounds__(256) void edgepool_self_kernel(
    const float* __restrict__ x,
    long long tiles,
    const float* __restrict__ W_t,
    const float* __restrict__ b_t,
    float* __restrict__ x_self)
{
    __shared__ _Float16 sW[OUT_CH * LDSW];
    load_Wt_to_lds(sW, W_t);
    __syncthreads();

    const int wave = threadIdx.x >> 5;
    const long long tile = (long long)blockIdx.x * 8 + wave;
    if (tile >= tiles) return;

    tile_compute<true>(x, nullptr, nullptr, b_t, nullptr, 0.0f, sW,
                       x_self, nullptr, tile * 16);
}

// ---------------------------------------------------------------------------
extern "C" void kernel_launch(void* const* d_in, const int* in_sizes, int n_in,
                              void* d_out, int out_size, void* d_ws, size_t ws_size,
                              hipStream_t stream) {
    (void)n_in; (void)d_ws; (void)ws_size; (void)out_size;

    const float* x   = (const float*)d_in[0];
    const int*   ei  = (const int*)d_in[1];   // edge_index [2,E] (int)
    // d_in[2] = batch (unused)
    const float* W_t = (const float*)d_in[3];
    const float* b_t = (const float*)d_in[4];
    const float* W_s = (const float*)d_in[5];
    const float* b_s = (const float*)d_in[6];

    const long long E = (long long)in_sizes[1] / 2;
    const long long N = (long long)in_sizes[0] / IN_CH;

    float* out      = (float*)d_out;
    float* x_merged = out;                               // [E, 128]
    float* x_self   = out + (size_t)E * OUT_CH;          // [N, 128]
    float* scores   = x_self + (size_t)N * OUT_CH;       // [E]

    const long long eTiles = (E + 15) / 16;
    const long long nTiles = (N + 15) / 16;
    const unsigned eBlocks = (unsigned)((eTiles + 7) / 8);
    const unsigned nBlocks = (unsigned)((nTiles + 7) / 8);

    edgepool_edge_kernel<<<dim3(eBlocks), dim3(256), 0, stream>>>(
        x, ei, E, eTiles, W_t, b_t, W_s, b_s, x_merged, scores);

    edgepool_self_kernel<<<dim3(nBlocks), dim3(256), 0, stream>>>(
        x, nTiles, W_t, b_t, x_self);
}
